// RunoffProducingCell_85478439125664
// MI455X (gfx1250) — compile-verified, matrix operand order checked by reading, not run on hardware
//
#include <hip/hip_runtime.h>
#include <hip/hip_bf16.h>

typedef __attribute__((ext_vector_type(16))) _Float16 v16h;
typedef __attribute__((ext_vector_type(8)))  _Float16 v8h;
typedef __attribute__((ext_vector_type(8)))  float    v8f;

#define ROWS 32      // batch rows per workgroup
#define XS   264     // f16 LDS row stride (halves), 528B, 16B aligned, non-pow2
#define FS   260     // f32 LDS row stride (floats), 1040B, 16B aligned

// ---- dynamic LDS layout (bytes) ----
#define HALF_ELEMS  (3 * ROWS * XS)                    // x0p, xcur, evr
#define FLOAT_ELEMS (3*ROWS*FS /*x0,hst,acc*/ + 256 /*wc*/ + 4*256 /*biases*/ \
                     + 4*ROWS /*pr,pin,pe,h256*/ + 256 /*red*/ + 2*ROWS /*rmax,rsum*/ \
                     + 2*257 /*c,r*/ + 256 /*we2last*/)
#define SMEM_BYTES  (HALF_ELEMS*2 + FLOAT_ELEMS*4)

// ======================= prep kernels =======================

// Wt[n][k] = (f16) W[k][n]   (256x256; for We2 uses rows k<256 of the 257x256)
__global__ void pack_wt_f16(const float* __restrict__ W, _Float16* __restrict__ Wt) {
    int idx = blockIdx.x * blockDim.x + threadIdx.x;   // 0..65535
    int n = idx >> 8, k = idx & 255;
    Wt[n * 256 + k] = (_Float16)W[k * 256 + n];
}

// c[j] = be1[j] + sum_k wc[k]*We1[k][j];  r[j] = We1[256][j];  we2last[n] = We2[256][n]
__global__ void prep_scalars(const float* __restrict__ wc,
                             const float* __restrict__ We1,
                             const float* __restrict__ be1,
                             const float* __restrict__ We2,
                             float* __restrict__ cvec, float* __restrict__ rvec,
                             float* __restrict__ we2last) {
    int j = threadIdx.x;
    if (j < 256) we2last[j] = We2[256 * 256 + j];
    if (j < 257) {
        float acc = be1[j];
        for (int k = 0; k < 256; ++k) acc = fmaf(wc[k], We1[k * 257 + j], acc);
        cvec[j] = acc;
        rvec[j] = We1[256 * 257 + j];
    }
}

// ======================= device helpers =======================

// ISA 16-bit A layout (16x32): lane<16 -> K 0..7 & 16..23 ; lane>=16 -> K 8..15 & 24..31
__device__ __forceinline__ v16h load_a_frag(const _Float16* __restrict__ xsrc,
                                            int lane, int mt, int kc) {
    int row = mt * 16 + (lane & 15);
    int kb  = kc * 32 + ((lane & 16) ? 8 : 0);
    const _Float16* p = xsrc + row * XS + kb;
    v8h lo = *(const v8h*)(p);
    v8h hi = *(const v8h*)(p + 16);
    return __builtin_shufflevector(lo, hi, 0,1,2,3,4,5,6,7,8,9,10,11,12,13,14,15);
}

// B fragment from pre-transposed Wt[n][k] (global, 512B row stride): mirror of A layout
__device__ __forceinline__ v16h load_b_frag(const _Float16* __restrict__ wt,
                                            int lane, int nt, int kc) {
    int n  = nt * 16 + (lane & 15);
    int kb = kc * 32 + ((lane & 16) ? 8 : 0);
    const _Float16* p = wt + n * 256 + kb;
    v8h lo = *(const v8h*)(p);
    v8h hi = *(const v8h*)(p + 16);
    return __builtin_shufflevector(lo, hi, 0,1,2,3,4,5,6,7,8,9,10,11,12,13,14,15);
}

// one [32x256] x [256x256] tile-GEMM; wave owns (mt, 4 n-tiles) -> 32 WMMAs
__device__ __forceinline__ void gemm_tile(const _Float16* __restrict__ xsrc,
                                          const _Float16* __restrict__ wt,
                                          int lane, int mt, int nq, v8f acc4[4]) {
    #pragma unroll
    for (int j = 0; j < 4; ++j) {
        v8f z; 
        #pragma unroll
        for (int e = 0; e < 8; ++e) z[e] = 0.0f;
        acc4[j] = z;
    }
    for (int kc = 0; kc < 8; ++kc) {
        v16h a = load_a_frag(xsrc, lane, mt, kc);
        #pragma unroll
        for (int j = 0; j < 4; ++j) {
            v16h b = load_b_frag(wt, lane, nq * 4 + j, kc);
            acc4[j] = __builtin_amdgcn_wmma_f32_16x16x32_f16(
                false, a, false, b, (short)0, acc4[j], false, false);
        }
    }
}

// two tanh steps of a mix head; stores tanh(h)+p as f16 next-activations
__device__ __forceinline__ void tanh_steps(const _Float16* __restrict__ x0p,
                                           _Float16* __restrict__ xcur,
                                           const _Float16* __restrict__ wt,
                                           const float* __restrict__ bvec,
                                           const float* __restrict__ pin, int t) {
    int lane = t & 31, wave = t >> 5, mt = wave >> 2, nq = wave & 3;
    for (int step = 0; step < 2; ++step) {
        const _Float16* src = step ? xcur : x0p;
        v8f acc4[4];
        gemm_tile(src, wt, lane, mt, nq, acc4);
        __syncthreads();                       // all reads of src done before overwrite
        #pragma unroll
        for (int j = 0; j < 4; ++j) {
            int col = (nq * 4 + j) * 16 + (lane & 15);
            #pragma unroll
            for (int rr = 0; rr < 8; ++rr) {
                int row = mt * 16 + ((lane & 16) ? rr + 8 : rr);
                float hv = acc4[j][rr] + bvec[col];
                xcur[row * XS + col] = (_Float16)(tanhf(hv) + pin[row]);
            }
        }
        __syncthreads();
    }
}

__device__ __forceinline__ void store_h(const v8f* acc4, const float* __restrict__ bvec,
                                        float* __restrict__ hst, int lane, int mt, int nq) {
    #pragma unroll
    for (int j = 0; j < 4; ++j) {
        int col = (nq * 4 + j) * 16 + (lane & 15);
        #pragma unroll
        for (int rr = 0; rr < 8; ++rr) {
            int row = mt * 16 + ((lane & 16) ? rr + 8 : rr);
            hst[row * FS + col] = acc4[j][rr] + bvec[col];
        }
    }
}

// row-wise softmax over hst[ROWS][256]; mode 0: acc=ratio*precip  1: evr=ratio
// 2: acc -= ratio*PE*evr    (8 threads per row, 32 cols each)
__device__ __forceinline__ void softmax_consume(float* __restrict__ hst,
                                                float* __restrict__ red,
                                                float* __restrict__ rmax,
                                                float* __restrict__ rsum,
                                                float* __restrict__ accb,
                                                _Float16* __restrict__ evr,
                                                const float* __restrict__ pr,
                                                const float* __restrict__ pe,
                                                int mode, int t) {
    int srow = t >> 3, seg = t & 7;
    const float* hrow = hst + srow * FS;
    float mx = -3.4e38f;
    for (int i = 0; i < 32; ++i) mx = fmaxf(mx, hrow[seg * 32 + i]);
    red[t] = mx;
    __syncthreads();
    if (t < ROWS) {
        float m = red[t * 8];
        for (int k = 1; k < 8; ++k) m = fmaxf(m, red[t * 8 + k]);
        rmax[t] = m;
    }
    __syncthreads();
    float m = rmax[srow], s = 0.0f;
    for (int i = 0; i < 32; ++i) s += __expf(hrow[seg * 32 + i] - m);
    red[t] = s;
    __syncthreads();
    if (t < ROWS) {
        float sum = 0.0f;
        for (int k = 0; k < 8; ++k) sum += red[t * 8 + k];
        rsum[t] = sum;
    }
    __syncthreads();
    float inv = 1.0f / rsum[srow];
    if (mode == 0) {
        float pv = pr[srow];
        for (int i = 0; i < 32; ++i) { int c = seg * 32 + i;
            accb[srow * FS + c] = __expf(hrow[c] - m) * inv * pv; }
    } else if (mode == 1) {
        for (int i = 0; i < 32; ++i) { int c = seg * 32 + i;
            evr[srow * XS + c] = (_Float16)(__expf(hrow[c] - m) * inv); }
    } else {
        float pv = pe[srow];
        for (int i = 0; i < 32; ++i) { int c = seg * 32 + i;
            accb[srow * FS + c] -= __expf(hrow[c] - m) * inv * pv * (float)evr[srow * XS + c]; }
    }
    __syncthreads();
}

// ======================= main fused kernel =======================
__global__ void __launch_bounds__(256, 1)
runoff_main(const float* __restrict__ lswr, const float* __restrict__ precip,
            const float* __restrict__ pevap, const float* __restrict__ wc,
            const float* __restrict__ b_wr, const float* __restrict__ b_inf,
            const float* __restrict__ b_ev, const float* __restrict__ be2,
            const _Float16* __restrict__ Wwr_t, const _Float16* __restrict__ Winf_t,
            const _Float16* __restrict__ Wev_t, const _Float16* __restrict__ We2_t,
            const float* __restrict__ we2last, const float* __restrict__ cvec,
            const float* __restrict__ rvec, float* __restrict__ out, int Btot) {
    extern __shared__ __align__(16) char smem_raw[];
    _Float16* x0p  = (_Float16*)smem_raw;            // [ROWS][XS] f16(x0 + p)
    _Float16* xcur = x0p + ROWS * XS;                // [ROWS][XS] current activations(+p)
    _Float16* evr  = xcur + ROWS * XS;               // [ROWS][XS] ev_ratio
    float* x0   = (float*)(evr + ROWS * XS);         // [ROWS][FS] last_soil_water (f32)
    float* hst  = x0 + ROWS * FS;                    // [ROWS][FS] logits staging
    float* accb = hst + ROWS * FS;                   // [ROWS][FS] runoff accumulator
    float* wcs  = accb + ROWS * FS;                  // 256
    float* bias = wcs + 256;                         // 4*256: wr, inf, ev, be2
    float* pr   = bias + 4 * 256;                    // ROWS raw precip
    float* pin  = pr + ROWS;                         // ROWS precip/48
    float* pe   = pin + ROWS;                        // ROWS potential evap
    float* h256 = pe + ROWS;                         // ROWS relu(c256+PE*r256)
    float* red  = h256 + ROWS;                       // 256
    float* rmax = red + 256;                         // ROWS
    float* rsum = rmax + ROWS;                       // ROWS
    float* cl   = rsum + ROWS;                       // 257
    float* rl   = cl + 257;                          // 257
    float* w2l  = rl + 257;                          // 256

    int t = threadIdx.x;
    int lane = t & 31, wave = t >> 5, mt = wave >> 2, nq = wave & 3;
    int b0 = blockIdx.x * ROWS;

    // ---- stage block-constant data ----
    if (t < 256) {
        wcs[t] = wc[t];
        bias[t]       = b_wr[t];
        bias[256 + t] = b_inf[t];
        bias[512 + t] = b_ev[t];
        bias[768 + t] = be2[t];
        w2l[t] = we2last[t];
        cl[t] = cvec[t]; rl[t] = rvec[t];
    }
    if (t == 0) { cl[256] = cvec[256]; rl[256] = rvec[256]; }
    if (t < ROWS) {
        int gb = min(b0 + t, Btot - 1);
        float pv = precip[gb];
        pr[t] = pv; pin[t] = pv * (1.0f / 48.0f);   // N*N_MIX = 48
        pe[t] = pevap[gb];
    }
    __syncthreads();
    // x0 = lswr * wc ; x0p = f16(x0 + p)
    for (int i = t; i < ROWS * 256; i += 256) {
        int row = i >> 8, col = i & 255;
        size_t gi = (size_t)min(b0 + row, Btot - 1) * 256 + col;
        float v = lswr[gi] * wcs[col];
        x0[row * FS + col]  = v;
        x0p[row * XS + col] = (_Float16)(v + pin[row]);
    }
    if (t < ROWS) h256[t] = fmaxf(cl[256] + pe[t] * rl[256], 0.0f);
    __syncthreads();

    // ================= head: infiltration (softmax) =================
    tanh_steps(x0p, xcur, Winf_t, bias + 256, pin, t);
    {
        v8f a4[4];
        gemm_tile(xcur, Winf_t, lane, mt, nq, a4);
        __syncthreads();
        store_h(a4, bias + 256, hst, lane, mt, nq);
        __syncthreads();
        softmax_consume(hst, red, rmax, rsum, accb, evr, pr, pe, /*mode=*/0, t);
    }

    // ================= head: evaporation ratio (softmax) =================
    tanh_steps(x0p, xcur, Wev_t, bias + 512, pin, t);
    {
        v8f a4[4];
        gemm_tile(xcur, Wev_t, lane, mt, nq, a4);
        __syncthreads();
        store_h(a4, bias + 512, hst, lane, mt, nq);
        __syncthreads();
        softmax_consume(hst, red, rmax, rsum, accb, evr, pr, pe, /*mode=*/1, t);
    }

    // ================= PE distribution MLP (rank-collapsed layer-1) =================
    {
        int srow = t >> 3, seg = t & 7;
        float pv = pe[srow];
        for (int i = 0; i < 32; ++i) { int c = seg * 32 + i;
            xcur[srow * XS + c] = (_Float16)fmaxf(cl[c] + pv * rl[c], 0.0f); }
        __syncthreads();
        v8f a4[4];
        gemm_tile(xcur, We2_t, lane, mt, nq, a4);
        __syncthreads();
        #pragma unroll
        for (int j = 0; j < 4; ++j) {
            int col = (nq * 4 + j) * 16 + (lane & 15);
            #pragma unroll
            for (int rr = 0; rr < 8; ++rr) {
                int row = mt * 16 + ((lane & 16) ? rr + 8 : rr);
                hst[row * FS + col] = a4[j][rr] + bias[768 + col] + h256[row] * w2l[col];
            }
        }
        __syncthreads();
        softmax_consume(hst, red, rmax, rsum, accb, evr, pr, pe, /*mode=*/2, t);
    }

    // ================= head: soil water ratio (sigmoid) + final combine =================
    tanh_steps(x0p, xcur, Wwr_t, bias, pin, t);
    {
        v8f a4[4];
        gemm_tile(xcur, Wwr_t, lane, mt, nq, a4);
        __syncthreads();
        #pragma unroll
        for (int j = 0; j < 4; ++j) {
            int col = (nq * 4 + j) * 16 + (lane & 15);
            #pragma unroll
            for (int rr = 0; rr < 8; ++rr) {
                int row = mt * 16 + ((lane & 16) ? rr + 8 : rr);
                float hv  = a4[j][rr] + bias[col];
                float swr = 1.0f / (1.0f + __expf(-hv));
                float run = accb[row * FS + col] - (swr * wcs[col] - x0[row * FS + col]);
                if (b0 + row < Btot) {
                    size_t gi = (size_t)(b0 + row) * 256 + col;
                    out[gi] = run;                               // runoff
                    out[(size_t)Btot * 256 + gi] = swr;          // soil_water_ratio
                }
            }
        }
    }
}

// ======================= host launcher =======================
extern "C" void kernel_launch(void* const* d_in, const int* in_sizes, int n_in,
                              void* d_out, int out_size, void* d_ws, size_t ws_size,
                              hipStream_t stream) {
    const float* lswr  = (const float*)d_in[0];
    const float* prec  = (const float*)d_in[1];
    const float* pev   = (const float*)d_in[2];
    const float* wc    = (const float*)d_in[3];
    const float* W_wr  = (const float*)d_in[4];
    const float* b_wr  = (const float*)d_in[5];
    const float* W_inf = (const float*)d_in[6];
    const float* b_inf = (const float*)d_in[7];
    const float* W_ev  = (const float*)d_in[8];
    const float* b_ev  = (const float*)d_in[9];
    const float* We1   = (const float*)d_in[10];
    const float* be1   = (const float*)d_in[11];
    const float* We2   = (const float*)d_in[12];
    const float* be2   = (const float*)d_in[13];
    int Btot = in_sizes[1];

    char* ws = (char*)d_ws;
    _Float16* Wwr_t  = (_Float16*)ws; ws += 256 * 256 * 2;
    _Float16* Winf_t = (_Float16*)ws; ws += 256 * 256 * 2;
    _Float16* Wev_t  = (_Float16*)ws; ws += 256 * 256 * 2;
    _Float16* We2_t  = (_Float16*)ws; ws += 256 * 256 * 2;
    float* we2last   = (float*)ws;    ws += 256 * 4;
    float* cvec      = (float*)ws;    ws += 260 * 4;
    float* rvec      = (float*)ws;    ws += 260 * 4;

    pack_wt_f16<<<256, 256, 0, stream>>>(W_wr,  Wwr_t);
    pack_wt_f16<<<256, 256, 0, stream>>>(W_inf, Winf_t);
    pack_wt_f16<<<256, 256, 0, stream>>>(W_ev,  Wev_t);
    pack_wt_f16<<<256, 256, 0, stream>>>(We2,   We2_t);
    prep_scalars<<<1, 288, 0, stream>>>(wc, We1, be1, We2, cvec, rvec, we2last);

    int grid = (Btot + ROWS - 1) / ROWS;
    runoff_main<<<grid, 256, SMEM_BYTES, stream>>>(
        lswr, prec, pev, wc, b_wr, b_inf, b_ev, be2,
        Wwr_t, Winf_t, Wev_t, We2_t, we2last, cvec, rvec,
        (float*)d_out, Btot);
}
